// TorchLinearGather_18150531793003
// MI455X (gfx1250) — matrix operand order repaired and checked by reading, not compiled, not run on hardware
//
#include <hip/hip_runtime.h>

// Problem constants (match the JAX reference)
#define BDIM 4
#define TDIM 4096
#define IDIM 2048
#define EDIM 16
#define JDIM 128
#define CDIM 1024

#define MROWS 64        // gathered token rows per workgroup
#define KCH   32        // K chunk per staged LDS tile (one bf16 WMMA K-step)
#define A_STR 40        // LDS row stride (bf16 elems) for A: 32 + 8 pad, 16B aligned
#define B_STR 136       // LDS row stride (bf16 elems) for Bt: 128 + 8 pad, 16B aligned

typedef __attribute__((ext_vector_type(16))) __bf16 v16bf;
typedef __attribute__((ext_vector_type(8)))  __bf16 v8bf;
typedef __attribute__((ext_vector_type(8)))  float  v8f;
typedef __attribute__((ext_vector_type(4)))  unsigned int uint4v;
typedef __attribute__((ext_vector_type(2)))  unsigned int uint2v;

// Split two fp32 values into packed-bf16 hi and lo words.
// hi = truncate-to-bf16 (exact residual), lo = truncate residual to bf16.
// Packing via v_perm_b32: result = { bf16(x1), bf16(x0) } (little-endian n-order).
__device__ __forceinline__ void split2(float x0, float x1,
                                       unsigned& hiP, unsigned& loP) {
  unsigned u0 = __builtin_bit_cast(unsigned, x0);
  unsigned u1 = __builtin_bit_cast(unsigned, x1);
  hiP = __builtin_amdgcn_perm(u1, u0, 0x07060302u);   // {u1[31:16], u0[31:16]}
  float h0 = __builtin_bit_cast(float, u0 & 0xFFFF0000u);
  float h1 = __builtin_bit_cast(float, u1 & 0xFFFF0000u);
  unsigned l0 = __builtin_bit_cast(unsigned, x0 - h0); // exact
  unsigned l1 = __builtin_bit_cast(unsigned, x1 - h1); // exact
  loP = __builtin_amdgcn_perm(l1, l0, 0x07060302u);
}

__global__ __launch_bounds__(256)
void linear_gather_wmma(const float* __restrict__ X,
                        const long long* __restrict__ ind,
                        const float* __restrict__ W,
                        float* __restrict__ out) {
  // Double-buffered LDS staging.
  // A = gathered X rows (row-major, WMMA-A: lane=M, elems=K)
  // Bt = W chunk transposed (row=k, WMMA-B: lane=N, elems=K)
  __shared__ alignas(16) unsigned short sAhi[2 * MROWS * A_STR];
  __shared__ alignas(16) unsigned short sAlo[2 * MROWS * A_STR];
  __shared__ alignas(16) unsigned short sBhi[2 * KCH * B_STR];
  __shared__ alignas(16) unsigned short sBlo[2 * KCH * B_STR];
  __shared__ int rows[MROWS];

  const int t    = threadIdx.x;
  const int mblk = blockIdx.x;          // 0..15  (C/64)
  const int e    = blockIdx.y;          // 0..15
  const int b    = blockIdx.z;          // 0..3
  const int c0   = mblk * MROWS;

  if (t < MROWS) {
    rows[t] = (int)ind[((size_t)(b * EDIM + e)) * CDIM + c0 + t];
  }
  __syncthreads();

  // ---- loader roles ----
  // A: thread -> (row ra, 8-float segment sega)
  const int ra = t >> 2, sega = t & 3;
  const float* aptr = X + ((size_t)b * TDIM + rows[ra]) * IDIM + sega * 8;
  // B: thread -> k-pair kp (2 consecutive k) for two groups of 4 W rows (j-cols)
  const int kp  = t & 15;               // k = 2*kp, 2*kp+1
  const int rg0 = t >> 4;               // row-group 0..15 (second group = +16)
  const float* wbase = W + (size_t)e * JDIM * IDIM;
  const float* bptr[2][4];
  #pragma unroll
  for (int u = 0; u < 2; ++u)
    #pragma unroll
    for (int i = 0; i < 4; ++i)
      bptr[u][i] = wbase + (size_t)((rg0 + u * 16) * 4 + i) * IDIM + kp * 2;

  float4 aR[2];
  float2 bR[2][4];

  auto loadA = [&](int kb) {
    aR[0] = *(const float4*)(aptr + kb);
    aR[1] = *(const float4*)(aptr + kb + 4);
  };
  auto loadB = [&](int kb) {
    #pragma unroll
    for (int u = 0; u < 2; ++u)
      #pragma unroll
      for (int i = 0; i < 4; ++i)
        bR[u][i] = *(const float2*)(bptr[u][i] + kb);
  };

  auto storeA = [&](unsigned bufOff) {
    const float* xf = (const float*)&aR[0];
    uint4v hi, lo;
    #pragma unroll
    for (int i = 0; i < 4; ++i) {
      unsigned h, l;
      split2(xf[2 * i], xf[2 * i + 1], h, l);
      hi[i] = h; lo[i] = l;
    }
    const unsigned off = bufOff + ra * A_STR + sega * 8;
    *(uint4v*)&sAhi[off] = hi;
    *(uint4v*)&sAlo[off] = lo;
  };
  auto storeB = [&](unsigned bufOff) {
    #pragma unroll
    for (int u = 0; u < 2; ++u) {
      const int rg = rg0 + u * 16;      // 4 consecutive j-columns: rg*4..rg*4+3
      uint2v h0, l0, h1, l1;
      unsigned a, c;
      split2(bR[u][0].x, bR[u][1].x, a, c); h0[0] = a; l0[0] = c;
      split2(bR[u][2].x, bR[u][3].x, a, c); h0[1] = a; l0[1] = c;
      split2(bR[u][0].y, bR[u][1].y, a, c); h1[0] = a; l1[0] = c;
      split2(bR[u][2].y, bR[u][3].y, a, c); h1[1] = a; l1[1] = c;
      const unsigned o0 = bufOff + (kp * 2 + 0) * B_STR + rg * 4;
      const unsigned o1 = bufOff + (kp * 2 + 1) * B_STR + rg * 4;
      *(uint2v*)&sBhi[o0] = h0;
      *(uint2v*)&sBhi[o1] = h1;
      *(uint2v*)&sBlo[o0] = l0;
      *(uint2v*)&sBlo[o1] = l1;
    }
  };

  // ---- compute roles ----
  const int lane = t & 31;
  const int wv   = t >> 5;              // 8 waves
  const int wm   = wv & 3;              // 16-row group within the 64 rows
  const int wn   = wv >> 2;             // 64-col group within J=128
  const int am   = lane & 15;           // A: lane = M
  const int ag   = lane >> 4;           // A: K-half selector

  const unsigned aFragOff = (wm * 16 + am) * A_STR;
  const unsigned bFragOff = am + wn * 64;             // B: lane = N

  auto fragA = [](const unsigned short* p, int g) -> v16bf {
    // lane holds k = g*8 + e (e<8) and k = 16 + g*8 + (e-8)
    v8bf x0 = *(const v8bf*)(p + g * 8);
    v8bf x1 = *(const v8bf*)(p + 16 + g * 8);
    return __builtin_shufflevector(x0, x1, 0,1,2,3,4,5,6,7,8,9,10,11,12,13,14,15);
  };
  auto fragB = [](const unsigned short* p, int g) -> v16bf {
    // lane = n; holds k = g*8 + e (e<8) and k = 16 + g*8 + (e-8)
    v16bf r;
    #pragma unroll
    for (int s = 0; s < 16; ++s) {
      const int k = (s < 8) ? (g * 8 + s) : (16 + g * 8 + (s - 8));
      r[s] = __builtin_bit_cast(__bf16, p[k * B_STR]);
    }
    return r;
  };

  v8f acc[4] = {v8f{}, v8f{}, v8f{}, v8f{}};

  loadA(0);
  loadB(0);
  const int nk = IDIM / KCH;            // 64 K-chunks
  for (int kc = 0; kc < nk; ++kc) {
    const unsigned aBuf = (unsigned)(kc & 1) * (MROWS * A_STR);
    const unsigned bBuf = (unsigned)(kc & 1) * (KCH * B_STR);

    storeA(aBuf);
    storeB(bBuf);
    __syncthreads();                    // single barrier: double-buffered

    // prefetch next chunk's globals; latency hidden under the WMMA block
    const int kn = (kc + 1 < nk) ? (kc + 1) * KCH : kc * KCH;
    loadA(kn);
    loadB(kn);

    const unsigned short* aHiB = &sAhi[aBuf + aFragOff];
    const unsigned short* aLoB = &sAlo[aBuf + aFragOff];
    const unsigned short* bHiB = &sBhi[bBuf + bFragOff];
    const unsigned short* bLoB = &sBlo[bBuf + bFragOff];

    v16bf aHi = fragA(aHiB, ag);
    v16bf aLo = fragA(aLoB, ag);
    #pragma unroll
    for (int tile = 0; tile < 4; ++tile) {
      v16bf bHi = fragB(bHiB + tile * 16, ag);
      v16bf bLo = fragB(bLoB + tile * 16, ag);
      // fp32 via bf16 split: hi*hi + hi*lo + lo*hi (lo*lo ~2^-16, dropped)
      acc[tile] = __builtin_amdgcn_wmma_f32_16x16x32_bf16(
          false, aHi, false, bHi, (short)0, acc[tile], false, false);
      acc[tile] = __builtin_amdgcn_wmma_f32_16x16x32_bf16(
          false, aHi, false, bLo, (short)0, acc[tile], false, false);
      acc[tile] = __builtin_amdgcn_wmma_f32_16x16x32_bf16(
          false, aLo, false, bHi, (short)0, acc[tile], false, false);
    }
  }

  // ---- epilogue: C/D layout -> out[b,e,c,j] (coalesced along j) ----
  // VGPR v, lane l: M = v + 8*(l>>4), N = l&15
  size_t outBase = ((size_t)(b * EDIM + e) * CDIM + c0 + wm * 16 + (lane >> 4) * 8) * JDIM
                   + wn * 64 + (lane & 15);
  #pragma unroll
  for (int tile = 0; tile < 4; ++tile)
    #pragma unroll
    for (int v = 0; v < 8; ++v)
      out[outBase + (size_t)v * JDIM + tile * 16] = acc[tile][v];
}

extern "C" void kernel_launch(void* const* d_in, const int* in_sizes, int n_in,
                              void* d_out, int out_size, void* d_ws, size_t ws_size,
                              hipStream_t stream) {
  (void)in_sizes; (void)n_in; (void)out_size; (void)d_ws; (void)ws_size;
  const float*     X   = (const float*)d_in[0];
  const long long* ind = (const long long*)d_in[1];   // reference dtype is int64
  const float*     W   = (const float*)d_in[2];
  float*           out = (float*)d_out;

  dim3 grid(CDIM / MROWS, EDIM, BDIM);  // 16 x 16 x 4 = 1024 workgroups
  linear_gather_wmma<<<grid, 256, 0, stream>>>(X, ind, W, out);
}
